// PolicyNetwork_80049600463335
// MI455X (gfx1250) — matrix operand order repaired and checked by reading, not compiled, not run on hardware
//
#include <hip/hip_runtime.h>
#include <math.h>

// ---------------------------------------------------------------------------
// Pointer-network (enc LSTM -> dec LSTM + attention + masked logits) for
// MI455X / gfx1250.
//   * All matmuls: v_wmma_f32_16x16x32_bf16 (wave32, f32 accumulate).
//   * Recurrent GEMM: per-WG A-panel (16 x [D|H] bf16) staged into LDS with
//     gfx1250 async copies (global_load_async_to_lds_b128 + s_wait_asynccnt),
//     shared by all 4 waves; B operand streamed from L2-resident bf16 weights.
//   * Decoder x-gather handled branch-free via a zeroed dummy row.
//   * bf16 weights (~6MB) + encoder states (64MB) stay L2-resident (192MB L2),
//     so the 128 serial decoder steps stream attention from L2, not HBM.
// ---------------------------------------------------------------------------

typedef __bf16 bf16_t;
typedef __attribute__((ext_vector_type(16))) __bf16 v16bf;
typedef __attribute__((ext_vector_type(8)))  float  v8f;

#define B_   512
#define T_   128
#define D_   256
#define H_   512
#define H4_  2048
#define L_   128

#define AP_  776   // LDS A-panel row stride in bf16 (768 + 8 pad -> conflict-free)

// ---------------- workspace layout (bytes, all region sizes 256-aligned) ----
#define OFF_INBF   ((size_t)0)                          // bf16 inputs   [B,T,D]
#define OFF_ENCBF  (OFF_INBF  + (size_t)B_*T_*D_*2)     // bf16 enc h    [B,T,H]
#define OFF_WIHE   (OFF_ENCBF + (size_t)B_*T_*H_*2)     // bf16 enc Wih  [4H,D]
#define OFF_WHHE   (OFF_WIHE  + (size_t)H4_*D_*2)       // bf16 enc Whh  [4H,H]
#define OFF_WIHD   (OFF_WHHE  + (size_t)H4_*H_*2)       // bf16 dec Wih
#define OFF_WHHD   (OFF_WIHD  + (size_t)H4_*D_*2)       // bf16 dec Whh
#define OFF_WDB    (OFF_WHHD  + (size_t)H4_*H_*2)       // bf16 Wd       [L,H]
#define OFF_BIASE  (OFF_WDB   + (size_t)L_*H_*2)        // f32 bih+bhh enc [4H]
#define OFF_BIASD  (OFF_BIASE + (size_t)H4_*4)          // f32 bih+bhh dec [4H]
#define OFF_GATES  (OFF_BIASD + (size_t)H4_*4)          // f32 gates     [B,4H]
#define OFF_HENC   (OFF_GATES + (size_t)B_*H4_*4)       // bf16 h enc    [B,H]
#define OFF_CENC   (OFF_HENC  + (size_t)B_*H_*2)        // f32  c enc    [B,H]
#define OFF_HDECB  (OFF_CENC  + (size_t)B_*H_*4)        // bf16 h dec    [B,H]
#define OFF_HDECF  (OFF_HDECB + (size_t)B_*H_*2)        // f32  h dec    [B,H]
#define OFF_CDEC   (OFF_HDECF + (size_t)B_*H_*4)        // f32  c dec    [B,H]
#define OFF_CTXF   (OFF_CDEC  + (size_t)B_*H_*4)        // f32  ctx      [B,H]
#define OFF_CTXB   (OFF_CTXF  + (size_t)B_*H_*4)        // bf16 ctx      [B,H]
#define OFF_MASK   (OFF_CTXB  + (size_t)B_*H_*2)        // f32 mask      [B,L]
#define OFF_ZROW   (OFF_MASK  + (size_t)B_*L_*4)        // bf16 zero row [D]
#define OFF_DIX    (OFF_ZROW  + (size_t)D_*2)           // i32 dec gather[B,T]
#define OFF_END    (OFF_DIX   + (size_t)B_*T_*4)

// ---------------- fragment helpers ------------------------------------------
union FragU { v16bf v; uint4 u[2]; };
union U4bf { uint4 u; __bf16 h[8]; };

// A operand (16x32 bf16, M x K): lanes 0-15: e0..7=K[0..7], e8..15=K[16..23];
// lanes 16-31: +8. p = row base + k0 (K-contiguous), koff = 8*(lane>>4).
__device__ __forceinline__ v16bf ldfragA(const bf16_t* p, int koff) {
  FragU f;
  f.u[0] = *reinterpret_cast<const uint4*>(p + koff);
  f.u[1] = *reinterpret_cast<const uint4*>(p + 16 + koff);
  return f.v;
}

// B operand (32x16 bf16, K x N), column per lane from row-major W[n,k]:
// lanes 0-15: e0..7=K[0..7], e8..15=K[16..23]; lanes 16-31: +8 (matches A).
__device__ __forceinline__ v16bf ldfragB(const bf16_t* p, int hi) {
  FragU f;
  const bf16_t* q = p + (hi << 3);
  f.u[0] = *reinterpret_cast<const uint4*>(q);
  f.u[1] = *reinterpret_cast<const uint4*>(q + 16);
  return f.v;
}

// gfx1250 async global->LDS copy (16B), tracked by ASYNCcnt.
__device__ __forceinline__ void async_g2l_b128(const bf16_t* lds_dst,
                                               const bf16_t* gsrc) {
  unsigned ldsoff = (unsigned)(uintptr_t)lds_dst;   // addr[31:0] == LDS offset
  asm volatile("global_load_async_to_lds_b128 %0, %1, off"
               :: "v"(ldsoff), "v"(gsrc) : "memory");
}
__device__ __forceinline__ void wait_async0() {
  asm volatile("s_wait_asynccnt 0" ::: "memory");
}

__device__ __forceinline__ float sigm(float x) {
  return 1.0f / (1.0f + __expf(-x));
}

// ---------------- small utility kernels -------------------------------------
__global__ void pn_cvt_bf16(const float* __restrict__ s, bf16_t* __restrict__ d, int n) {
  int i = blockIdx.x * blockDim.x + threadIdx.x;
  if (i < n) d[i] = (bf16_t)s[i];
}

__global__ void pn_bias_sum(const float* __restrict__ a, const float* __restrict__ b,
                            float* __restrict__ o, int n) {
  int i = blockIdx.x * blockDim.x + threadIdx.x;
  if (i < n) o[i] = a[i] + b[i];
}

__global__ void pn_zero_u32(unsigned* __restrict__ p, int n) {
  int i = blockIdx.x * blockDim.x + threadIdx.x;
  if (i < n) p[i] = 0u;
}

// dix[b,0] = -1 (x0 = zeros); dix[b,t] = clip(target_indices[b,t-1], 0, T-1)
__global__ void pn_dix(const int* __restrict__ tidx, int* __restrict__ dix) {
  int j = blockIdx.x * blockDim.x + threadIdx.x;
  if (j >= B_ * T_) return;
  int t = j & (T_ - 1);
  if (t == 0) { dix[j] = -1; return; }
  int v = tidx[j - 1];
  v = v < 0 ? 0 : (v > T_ - 1 ? T_ - 1 : v);
  dix[j] = v;
}

// mask[b, dix[b,tnext]] = 1   (the index selected at step tnext-1)
__global__ void pn_mask_upd(float* __restrict__ mask, const int* __restrict__ dix, int tnext) {
  int b = blockIdx.x * blockDim.x + threadIdx.x;
  if (b >= B_) return;
  int ix = dix[b * T_ + tnext];
  if (ix >= 0) mask[(size_t)b * L_ + ix] = 1.0f;
}

// ---------------- recurrent gates GEMM (WMMA bf16, async LDS A-panel) --------
// gates[b,n] = sum_k x_t[b,k]*Wih[n,k] + sum_k h[b,k]*Whh[n,k] + bias[n]
// M=B=512, N=4H=2048, K=D+H=768. Wave tile 16x64 (4 accumulators); block =
// 4 waves = 16x256 tile sharing one LDS A-panel; grid (32, 8).
__global__ __launch_bounds__(128)
void pn_rec_gemm(const bf16_t* __restrict__ xin, const int* __restrict__ dix,
                 const bf16_t* __restrict__ zrow, int mode, int t,
                 const bf16_t* __restrict__ hprev,
                 const bf16_t* __restrict__ Wih, const bf16_t* __restrict__ Whh,
                 const float* __restrict__ bias, float* __restrict__ gates) {
  __shared__ bf16_t shA[16][AP_];   // 16 rows x (256 x-part | 512 h-part), padded

  const int tid  = threadIdx.x;
  const int lane = tid & 31;
  const int wave = tid >> 5;
  const int l15  = lane & 15;
  const int hi   = lane >> 4;
  const int koff = hi << 3;
  const int m0 = blockIdx.x << 4;
  const int n0 = (blockIdx.y << 8) + (wave << 6);

  // ---- async-stage the A-panel: x rows (gathered for decoder) then h rows --
  // x part: 16 rows * 32 16B-chunks = 512 chunks
  for (int c = tid; c < 512; c += 128) {
    const int r  = c >> 5;
    const int ck = (c & 31) << 3;             // bf16 offset within row
    const bf16_t* src;
    if (mode == 0) {
      src = xin + ((size_t)(m0 + r) * T_ + t) * D_ + ck;
    } else {
      const int ix = dix[(m0 + r) * T_ + t];
      src = (ix < 0) ? (zrow + ck)
                     : (xin + ((size_t)(m0 + r) * T_ + ix) * D_ + ck);
    }
    async_g2l_b128(&shA[r][ck], src);
  }
  // h part: 16 rows * 64 16B-chunks = 1024 chunks
  for (int c = tid; c < 1024; c += 128) {
    const int r  = c >> 6;
    const int ck = (c & 63) << 3;
    async_g2l_b128(&shA[r][D_ + ck], hprev + (size_t)(m0 + r) * H_ + ck);
  }
  wait_async0();
  __syncthreads();

  const bf16_t* wx0 = Wih + (size_t)(n0 +  0 + l15) * D_;
  const bf16_t* wx1 = Wih + (size_t)(n0 + 16 + l15) * D_;
  const bf16_t* wx2 = Wih + (size_t)(n0 + 32 + l15) * D_;
  const bf16_t* wx3 = Wih + (size_t)(n0 + 48 + l15) * D_;
  const bf16_t* wh0 = Whh + (size_t)(n0 +  0 + l15) * H_;
  const bf16_t* wh1 = Whh + (size_t)(n0 + 16 + l15) * H_;
  const bf16_t* wh2 = Whh + (size_t)(n0 + 32 + l15) * H_;
  const bf16_t* wh3 = Whh + (size_t)(n0 + 48 + l15) * H_;

  v8f a0 = {}, a1 = {}, a2 = {}, a3 = {};

  // x part: K = 0..D
  for (int k0 = 0; k0 < D_; k0 += 32) {
    v16bf af = ldfragA(&shA[l15][k0], koff);
    v16bf b0 = ldfragB(wx0 + k0, hi);
    v16bf b1 = ldfragB(wx1 + k0, hi);
    v16bf b2 = ldfragB(wx2 + k0, hi);
    v16bf b3 = ldfragB(wx3 + k0, hi);
    a0 = __builtin_amdgcn_wmma_f32_16x16x32_bf16(false, af, false, b0, (short)0, a0, false, false);
    a1 = __builtin_amdgcn_wmma_f32_16x16x32_bf16(false, af, false, b1, (short)0, a1, false, false);
    a2 = __builtin_amdgcn_wmma_f32_16x16x32_bf16(false, af, false, b2, (short)0, a2, false, false);
    a3 = __builtin_amdgcn_wmma_f32_16x16x32_bf16(false, af, false, b3, (short)0, a3, false, false);
  }
  // h part: K = D..D+H (the serial-dependency operand)
  for (int k0 = 0; k0 < H_; k0 += 32) {
    v16bf af = ldfragA(&shA[l15][D_ + k0], koff);
    v16bf b0 = ldfragB(wh0 + k0, hi);
    v16bf b1 = ldfragB(wh1 + k0, hi);
    v16bf b2 = ldfragB(wh2 + k0, hi);
    v16bf b3 = ldfragB(wh3 + k0, hi);
    a0 = __builtin_amdgcn_wmma_f32_16x16x32_bf16(false, af, false, b0, (short)0, a0, false, false);
    a1 = __builtin_amdgcn_wmma_f32_16x16x32_bf16(false, af, false, b1, (short)0, a1, false, false);
    a2 = __builtin_amdgcn_wmma_f32_16x16x32_bf16(false, af, false, b2, (short)0, a2, false, false);
    a3 = __builtin_amdgcn_wmma_f32_16x16x32_bf16(false, af, false, b3, (short)0, a3, false, false);
  }

  const float bv0 = bias[n0 +  0 + l15];
  const float bv1 = bias[n0 + 16 + l15];
  const float bv2 = bias[n0 + 32 + l15];
  const float bv3 = bias[n0 + 48 + l15];
#pragma unroll
  for (int r = 0; r < 8; ++r) {
    float* gr = gates + (size_t)(m0 + (hi << 3) + r) * H4_;
    gr[n0 +  0 + l15] = a0[r] + bv0;
    gr[n0 + 16 + l15] = a1[r] + bv1;
    gr[n0 + 32 + l15] = a2[r] + bv2;
    gr[n0 + 48 + l15] = a3[r] + bv3;
  }
}

// ---------------- LSTM pointwise cell ---------------------------------------
__global__ void pn_lstm_cell(const float* __restrict__ gates, float* __restrict__ c,
                             bf16_t* __restrict__ hb, float* __restrict__ hf,
                             bf16_t* __restrict__ encout, int t) {
  int j = blockIdx.x * blockDim.x + threadIdx.x;
  if (j >= B_ * H_) return;
  int b = j >> 9;              // H_ == 512
  int k = j & (H_ - 1);
  const float* g = gates + (size_t)b * H4_;
  float gi = g[k];
  float gf = g[H_ + k];
  float gg = g[2 * H_ + k];
  float go = g[3 * H_ + k];
  float cn = sigm(gf) * c[j] + sigm(gi) * tanhf(gg);
  float hn = sigm(go) * tanhf(cn);
  c[j]  = cn;
  hb[j] = (bf16_t)hn;
  if (hf)     hf[j] = hn;
  if (encout) encout[((size_t)b * T_ + t) * H_ + k] = (bf16_t)hn;
}

// ---------------- attention + rot head (one WG per batch row) ---------------
__global__ __launch_bounds__(128)
void pn_attn(const float* __restrict__ hdec, const bf16_t* __restrict__ enc,
             const float* __restrict__ Wr, const float* __restrict__ br,
             float* __restrict__ ctxf, bf16_t* __restrict__ ctxb,
             float* __restrict__ out_rot, int t) {
  __shared__ float sh[H_];     // h_dec[b], then reused for ctx
  __shared__ float sc[T_];     // attn weights
  __shared__ float red[T_];    // reduction scratch
  const int b = blockIdx.x;
  const int tid = threadIdx.x; // 128 threads = T_

  for (int k = tid; k < H_; k += 128) sh[k] = hdec[(size_t)b * H_ + k];
  __syncthreads();

  // scores[b, tid] = h . enc[b, tid, :]   (bf16, vectorized 8-wide)
  const uint4* e4 = reinterpret_cast<const uint4*>(enc + ((size_t)b * T_ + tid) * H_);
  float s = 0.0f;
  for (int c = 0; c < H_ / 8; ++c) {
    U4bf w; w.u = e4[c];
#pragma unroll
    for (int i = 0; i < 8; ++i) s += sh[c * 8 + i] * (float)w.h[i];
  }
  red[tid] = s;
  __syncthreads();
  for (int off = 64; off > 0; off >>= 1) {
    if (tid < off) red[tid] = fmaxf(red[tid], red[tid + off]);
    __syncthreads();
  }
  const float mx = red[0];
  __syncthreads();
  float e = __expf(s - mx);
  red[tid] = e;
  __syncthreads();
  for (int off = 64; off > 0; off >>= 1) {
    if (tid < off) red[tid] += red[tid + off];
    __syncthreads();
  }
  const float inv = 1.0f / red[0];
  sc[tid] = e * inv;
  __syncthreads();

  // ctx[b,k] = sum_t attn[t] * enc[b,t,k]   (lane-coalesced across k)
  for (int k = tid; k < H_; k += 128) {
    float acc = 0.0f;
    for (int tt = 0; tt < T_; ++tt)
      acc += sc[tt] * (float)enc[((size_t)b * T_ + tt) * H_ + k];
    ctxf[(size_t)b * H_ + k] = acc;
    ctxb[(size_t)b * H_ + k] = (bf16_t)acc;
    sh[k] = acc;
  }
  __syncthreads();

  // rot = ctx @ Wr.T + br  (4 outputs)
  if (tid < 4) {
    float r = br[tid];
    const float* wr = Wr + (size_t)tid * H_;
    for (int k = 0; k < H_; ++k) r += wr[k] * sh[k];
    out_rot[((size_t)b * T_ + t) * 4 + tid] = r;
  }
}

// ---------------- masked logits GEMM (WMMA bf16) ----------------------------
// logits[b,n] = ctx[b,:]·Wd[n,:] + bd[n], masked to -inf. M=512,N=128,K=512.
__global__ __launch_bounds__(128)
void pn_logits(const bf16_t* __restrict__ ctxb, const bf16_t* __restrict__ Wdb,
               const float* __restrict__ bd, const float* __restrict__ mask,
               float* __restrict__ out, int t) {
  const int lane = threadIdx.x & 31;
  const int wave = threadIdx.x >> 5;
  const int l15  = lane & 15;
  const int hi   = lane >> 4;
  const int koff = hi << 3;
  const int m0 = (blockIdx.x << 5) + ((wave >> 1) << 4);
  const int n0 = (wave & 1) << 6;

  const bf16_t* arow = ctxb + (size_t)(m0 + l15) * H_;
  const bf16_t* w0 = Wdb + (size_t)(n0 +  0 + l15) * H_;
  const bf16_t* w1 = Wdb + (size_t)(n0 + 16 + l15) * H_;
  const bf16_t* w2 = Wdb + (size_t)(n0 + 32 + l15) * H_;
  const bf16_t* w3 = Wdb + (size_t)(n0 + 48 + l15) * H_;

  v8f a0 = {}, a1 = {}, a2 = {}, a3 = {};
  for (int k0 = 0; k0 < H_; k0 += 32) {
    v16bf af = ldfragA(arow + k0, koff);
    v16bf b0 = ldfragB(w0 + k0, hi);
    v16bf b1 = ldfragB(w1 + k0, hi);
    v16bf b2 = ldfragB(w2 + k0, hi);
    v16bf b3 = ldfragB(w3 + k0, hi);
    a0 = __builtin_amdgcn_wmma_f32_16x16x32_bf16(false, af, false, b0, (short)0, a0, false, false);
    a1 = __builtin_amdgcn_wmma_f32_16x16x32_bf16(false, af, false, b1, (short)0, a1, false, false);
    a2 = __builtin_amdgcn_wmma_f32_16x16x32_bf16(false, af, false, b2, (short)0, a2, false, false);
    a3 = __builtin_amdgcn_wmma_f32_16x16x32_bf16(false, af, false, b3, (short)0, a3, false, false);
  }

  const float ninf = -__builtin_inff();
  const float bv0 = bd[n0 +  0 + l15];
  const float bv1 = bd[n0 + 16 + l15];
  const float bv2 = bd[n0 + 32 + l15];
  const float bv3 = bd[n0 + 48 + l15];
#pragma unroll
  for (int r = 0; r < 8; ++r) {
    const int row = m0 + (hi << 3) + r;         // batch index
    const float* mrow = mask + (size_t)row * L_;
    float* orow = out + ((size_t)row * T_ + t) * L_;
    float v0 = a0[r] + bv0; if (mrow[n0 +  0 + l15] > 0.0f) v0 = ninf;
    float v1 = a1[r] + bv1; if (mrow[n0 + 16 + l15] > 0.0f) v1 = ninf;
    float v2 = a2[r] + bv2; if (mrow[n0 + 32 + l15] > 0.0f) v2 = ninf;
    float v3 = a3[r] + bv3; if (mrow[n0 + 48 + l15] > 0.0f) v3 = ninf;
    orow[n0 +  0 + l15] = v0;
    orow[n0 + 16 + l15] = v1;
    orow[n0 + 32 + l15] = v2;
    orow[n0 + 48 + l15] = v3;
  }
}

// ---------------------------------------------------------------------------
extern "C" void kernel_launch(void* const* d_in, const int* in_sizes, int n_in,
                              void* d_out, int out_size, void* d_ws, size_t ws_size,
                              hipStream_t stream) {
  const float* inputs = (const float*)d_in[0];
  const int*   tidx   = (const int*)  d_in[1];
  const float* eWih   = (const float*)d_in[2];
  const float* eWhh   = (const float*)d_in[3];
  const float* ebih   = (const float*)d_in[4];
  const float* ebhh   = (const float*)d_in[5];
  const float* dWih   = (const float*)d_in[6];
  const float* dWhh   = (const float*)d_in[7];
  const float* dbih   = (const float*)d_in[8];
  const float* dbhh   = (const float*)d_in[9];
  const float* Wd     = (const float*)d_in[10];
  const float* bd     = (const float*)d_in[11];
  const float* Wr     = (const float*)d_in[12];
  const float* br     = (const float*)d_in[13];

  char* ws = (char*)d_ws;
  bf16_t* in_bf  = (bf16_t*)(ws + OFF_INBF);
  bf16_t* encbf  = (bf16_t*)(ws + OFF_ENCBF);
  bf16_t* wihe   = (bf16_t*)(ws + OFF_WIHE);
  bf16_t* whhe   = (bf16_t*)(ws + OFF_WHHE);
  bf16_t* wihd   = (bf16_t*)(ws + OFF_WIHD);
  bf16_t* whhd   = (bf16_t*)(ws + OFF_WHHD);
  bf16_t* wdb    = (bf16_t*)(ws + OFF_WDB);
  float*  biase  = (float*) (ws + OFF_BIASE);
  float*  biasd  = (float*) (ws + OFF_BIASD);
  float*  gates  = (float*) (ws + OFF_GATES);
  bf16_t* henc   = (bf16_t*)(ws + OFF_HENC);
  float*  cenc   = (float*) (ws + OFF_CENC);
  bf16_t* hdecb  = (bf16_t*)(ws + OFF_HDECB);
  float*  hdecf  = (float*) (ws + OFF_HDECF);
  float*  cdec   = (float*) (ws + OFF_CDEC);
  float*  ctxf   = (float*) (ws + OFF_CTXF);
  bf16_t* ctxb   = (bf16_t*)(ws + OFF_CTXB);
  float*  maskp  = (float*) (ws + OFF_MASK);
  bf16_t* zrow   = (bf16_t*)(ws + OFF_ZROW);
  int*    dix    = (int*)   (ws + OFF_DIX);

  float* out     = (float*)d_out;                      // logits [B,T,L]
  float* out_rot = out + (size_t)B_ * T_ * L_;         // rot    [B,T,4]

  const int TPB = 256;
  auto blocks = [](size_t n, int tpb) { return (unsigned)((n + tpb - 1) / tpb); };

  // --- setup: conversions, bias sums, gather indices, state zeroing ---------
  pn_cvt_bf16<<<blocks((size_t)B_*T_*D_, TPB), TPB, 0, stream>>>(inputs, in_bf, B_*T_*D_);
  pn_cvt_bf16<<<blocks((size_t)H4_*D_,  TPB), TPB, 0, stream>>>(eWih, wihe, H4_*D_);
  pn_cvt_bf16<<<blocks((size_t)H4_*H_,  TPB), TPB, 0, stream>>>(eWhh, whhe, H4_*H_);
  pn_cvt_bf16<<<blocks((size_t)H4_*D_,  TPB), TPB, 0, stream>>>(dWih, wihd, H4_*D_);
  pn_cvt_bf16<<<blocks((size_t)H4_*H_,  TPB), TPB, 0, stream>>>(dWhh, whhd, H4_*H_);
  pn_cvt_bf16<<<blocks((size_t)L_*H_,   TPB), TPB, 0, stream>>>(Wd,   wdb,  L_*H_);
  pn_bias_sum<<<blocks(H4_, TPB), TPB, 0, stream>>>(ebih, ebhh, biase, H4_);
  pn_bias_sum<<<blocks(H4_, TPB), TPB, 0, stream>>>(dbih, dbhh, biasd, H4_);
  pn_dix<<<blocks((size_t)B_*T_, TPB), TPB, 0, stream>>>(tidx, dix);
  {
    size_t zbytes = OFF_DIX - OFF_HENC;   // h/c enc+dec, ctx, mask, zero-row
    pn_zero_u32<<<blocks(zbytes / 4, TPB), TPB, 0, stream>>>(
        (unsigned*)(ws + OFF_HENC), (int)(zbytes / 4));
  }

  const dim3 gGemm(B_ / 16, H4_ / 256);   // (32, 8)
  const unsigned gCell = blocks((size_t)B_ * H_, TPB);

  // --- encoder: 128 serial steps -------------------------------------------
  for (int t = 0; t < T_; ++t) {
    pn_rec_gemm<<<gGemm, 128, 0, stream>>>(in_bf, dix, zrow, 0, t, henc,
                                           wihe, whhe, biase, gates);
    pn_lstm_cell<<<gCell, TPB, 0, stream>>>(gates, cenc, henc,
                                            (float*)nullptr, encbf, t);
  }

  // --- decoder: 128 serial steps (gemm -> cell -> attention -> logits) -----
  for (int t = 0; t < T_; ++t) {
    pn_rec_gemm<<<gGemm, 128, 0, stream>>>(in_bf, dix, zrow, 1, t, hdecb,
                                           wihd, whhd, biasd, gates);
    pn_lstm_cell<<<gCell, TPB, 0, stream>>>(gates, cdec, hdecb,
                                            hdecf, (bf16_t*)nullptr, t);
    pn_attn<<<B_, 128, 0, stream>>>(hdecf, encbf, Wr, br, ctxf, ctxb, out_rot, t);
    pn_logits<<<B_ / 32, 128, 0, stream>>>(ctxb, wdb, bd, maskp, out, t);
    if (t + 1 < T_)
      pn_mask_upd<<<blocks(B_, TPB), TPB, 0, stream>>>(maskp, dix, t + 1);
  }
}
